// pyyTest_90933047591178
// MI455X (gfx1250) — matrix operand order repaired and checked by reading, not compiled, not run on hardware
//
#include <hip/hip_runtime.h>
#include <math.h>

// ---------------------------------------------------------------------------
// CDNA5 (gfx1250, wave32) implementation of the Proxy-Anchor-style loss.
// Dominant HBM cost is the 204.8 MB P_f copy (~8.8us @ 23.3 TB/s).  The GEMM
// is reduced 10x by exploiting the pNum structure (only k=0 active + <=128
// inserted slots) and fused with the exp/mask/reduce epilogue so the 51 MB
// cos matrix is never materialized.  GEMM uses v_wmma_f32_16x16x32_f16 with
// global_load_async_to_lds_b128 staging for both tiles (ASYNCcnt-tracked).
// ---------------------------------------------------------------------------

typedef __attribute__((ext_vector_type(16))) _Float16 v16h;
typedef __attribute__((ext_vector_type(8)))  float    v8f;

#define BATCH 128
#define EMB   256
#define NCLS  10000
#define KPRX  10
#define ALPHA 32.0f
#define MRG   0.1f

// workspace byte offsets (all 16-aligned); total ~5.37 MB
#define WS_XN_F32 0          // 128*256*4     = 131072
#define WS_XN_F16 131072     // 128*256*2     =  65536
#define WS_PN_F16 196608     // 10000*256*2   = 5120000 (normalized k=0 rows)
#define WS_COST   5316608    // 128*10*4      =    5120 (cos at label columns)
#define WS_NSIM   5321728    // 10000*4       =   40000 (N_sim_sum per class)
#define WS_SCAN   5361728    // [0]=insCount [1]=posTerm [2..]=triples

// CDNA5 async-to-LDS copy: 16 bytes/lane, tracked by ASYNCcnt.
__device__ __forceinline__ void async_ld_b128(unsigned lds_byte_off, const void* gaddr) {
  asm volatile("global_load_async_to_lds_b128 %0, %1, off"
               :: "v"(lds_byte_off), "v"(gaddr)
               : "memory");
}
__device__ __forceinline__ void wait_async0() {
  asm volatile("s_wait_asynccnt 0x0" ::: "memory");
}

// --- 1. row-normalize X; keep f32 (for dots) and f16 (WMMA A matrix) -------
__global__ void k_normx(const float* __restrict__ X, float* __restrict__ XnF32,
                        _Float16* __restrict__ XnF16) {
  __shared__ float red[EMB];
  int b = blockIdx.x, t = threadIdx.x;
  float v = X[b * EMB + t];
  red[t] = v * v;
  __syncthreads();
  for (int s = EMB / 2; s > 0; s >>= 1) {
    if (t < s) red[t] += red[t + s];
    __syncthreads();
  }
  float nv = v * rsqrtf(red[0] + 1e-12f);
  XnF32[b * EMB + t] = nv;
  XnF16[b * EMB + t] = (_Float16)nv;
}

// --- 2. normalize k=0 proxy rows (rows 0..C-1) into f16 --------------------
__global__ void k_pnorm(const float* __restrict__ P, _Float16* __restrict__ PnF16) {
  int w = threadIdx.x >> 5, lane = threadIdx.x & 31;
  int r = blockIdx.x * 8 + w;
  if (r >= NCLS) return;
  const float* row = P + (size_t)r * EMB;
  float x[8]; float s = 0.f;
#pragma unroll
  for (int j = 0; j < 8; j++) { x[j] = row[lane + j * 32]; s += x[j] * x[j]; }
  for (int off = 16; off; off >>= 1) s += __shfl_xor(s, off, 32);
  float inv = rsqrtf(s + 1e-12f);
  _Float16* orow = PnF16 + (size_t)r * EMB;
#pragma unroll
  for (int j = 0; j < 8; j++) orow[lane + j * 32] = (_Float16)(x[j] * inv);
}

// --- 3. cos at label columns: 1280 wave-sized dot products (f32, all k) ----
__global__ void k_cost(const float* __restrict__ P, const float* __restrict__ XnF32,
                       const int* __restrict__ T, float* __restrict__ cosT) {
  int w = threadIdx.x >> 5, lane = threadIdx.x & 31;
  int g = blockIdx.x * 8 + w;              // 0..1279
  if (g >= BATCH * KPRX) return;
  int b = g / KPRX, k = g % KPRX;
  int r = k * NCLS + T[b];
  const float* pr = P + (size_t)r * EMB;
  const float* xr = XnF32 + b * EMB;
  float dpp = 0.f, dxp = 0.f;
#pragma unroll
  for (int j = 0; j < 8; j++) {
    int e = lane + j * 32;
    float p = pr[e];
    dpp += p * p;
    dxp += p * xr[e];
  }
  for (int off = 16; off; off >>= 1) {
    dpp += __shfl_xor(dpp, off, 32);
    dxp += __shfl_xor(dxp, off, 32);
  }
  if (lane == 0) cosT[b * KPRX + k] = dxp * rsqrtf(dpp + 1e-12f);
}

// --- 4. sequential scan (128 steps; only column T[i] is ever consumed) -----
__global__ void k_scan(const int* __restrict__ T, const int* __restrict__ pTotal,
                       const unsigned char* __restrict__ pNum,
                       const float* __restrict__ cosT,
                       int* scanI, float* scanF) {
  if (threadIdx.x != 0) return;
  int cls[BATCH]; int pt[BATCH]; unsigned int msk[BATCH]; float psum[BATCH];
  int nent = 0, insCnt = 0;
  for (int i = 0; i < BATCH; i++) {
    int lab = T[i];
    int e = -1;
    for (int q = 0; q < nent; q++) if (cls[q] == lab) { e = q; break; }
    if (e < 0) {
      e = nent++; cls[e] = lab; pt[e] = pTotal[lab]; psum[e] = 0.f;
      unsigned int m = 0;
      for (int k = 0; k < KPRX; k++) if (pNum[k * NCLS + lab]) m |= 1u << k;
      msk[e] = m;
    }
    float ps = 0.f, mx = -1e30f;
    for (int k = 0; k < KPRX; k++) if ((msk[e] >> k) & 1u) {
      float pe = __expf(-ALPHA * (cosT[i * KPRX + k] - MRG));
      ps += pe; if (pe > mx) mx = pe;
    }
    psum[e] += ps;                       // psim accumulated BEFORE insertion
    int idx = pt[e];
    if (mx > 0.99f && mx < 1.01f && idx < KPRX) {
      scanI[2 + insCnt * 3 + 0] = lab;
      scanI[2 + insCnt * 3 + 1] = idx;
      scanI[2 + insCnt * 3 + 2] = i;
      insCnt++;
      msk[e] |= 1u << idx;
      pt[e] = idx + 1;
    }
  }
  float pos = 0.f;
  for (int q = 0; q < nent; q++) pos += log1pf(psum[q]);
  pos /= (float)nent;                    // num_valid_proxies = #classes in batch
  scanI[0] = insCnt;
  scanF[1] = pos;
}

// --- 5. WMMA neg-GEMM over k=0 proxy rows with fused exp/mask epilogue -----
// Block: 256 threads (8 waves), tile M=128 (batch) x N=128 (classes), K=256.
// LDS (dynamic): A (Xn f16, 64KB) + B (Pn f16 tile, 64KB) + T (512B).
// Both tiles staged with global_load_async_to_lds_b128 (ASYNCcnt).
__global__ void __launch_bounds__(256)
k_neggemm(const _Float16* __restrict__ PnF16, const _Float16* __restrict__ XnF16,
          const int* __restrict__ T, const unsigned char* __restrict__ pNum0,
          float* __restrict__ Nsim) {
  extern __shared__ char smem[];
  _Float16* Ah = (_Float16*)smem;             // [128][256]
  _Float16* Bh = Ah + 128 * 256;              // [128][256] (row = class)
  int* Tl = (int*)(Bh + 128 * 256);
  const unsigned ldsBase = __builtin_amdgcn_groupstaticsize();  // dynamic LDS origin
  int t = threadIdx.x;
  int cBase = blockIdx.x * 128;

  if (t < BATCH) Tl[t] = T[t];
  // A tile: 4096 x 16B async copies into LDS [0, 64K)
  for (int i = t; i < 4096; i += 256)
    async_ld_b128(ldsBase + (unsigned)(i * 16), (const char*)XnF16 + i * 16);
  // B tile: row rr <- normalized f16 proxy row cBase+rr, LDS [64K, 128K)
  for (int i = t; i < 4096; i += 256) {
    int rr = i >> 5, ch = i & 31;             // 32 x 16B chunks per 512B row
    int r = cBase + rr;
    if (r < NCLS)
      async_ld_b128(ldsBase + (unsigned)(65536 + rr * 512 + ch * 16),
                    (const char*)PnF16 + (size_t)r * 512 + ch * 16);
  }
  wait_async0();
  __syncthreads();

  int wave = t >> 5, lane = t & 31, ln = lane & 15, kh = lane >> 4;
  v8f acc[8] = {};
  const _Float16* Bp = Bh + (wave * 16 + ln) * 256;   // this lane's class row
#pragma unroll
  for (int kt = 0; kt < 8; kt++) {                     // K = 8 x 32
    // B fragment (32x16): v0-3 -> K=kh*8..+7, v4-7 -> K=16+kh*8..+7
    union { v16h h; uint4 u[2]; } bf;
    bf.u[0] = *(const uint4*)(Bp + kt * 32 + kh * 8);
    bf.u[1] = *(const uint4*)(Bp + kt * 32 + kh * 8 + 16);
#pragma unroll
    for (int mt = 0; mt < 8; mt++) {                   // M = 8 x 16
      // A fragment (16x32): v0-3 -> K=kh*8..+7, v4-7 -> K=16+kh*8..+7
      const _Float16* Ap = Ah + (mt * 16 + ln) * 256 + kt * 32 + kh * 8;
      union { v16h h; uint4 u[2]; } af;
      af.u[0] = *(const uint4*)(Ap);
      af.u[1] = *(const uint4*)(Ap + 16);
      acc[mt] = __builtin_amdgcn_wmma_f32_16x16x32_f16(
          false, af.h, false, bf.h, (short)0, acc[mt], false, false);
    }
  }
  // epilogue: N_sim_sum[c] = sum_b (T[b]!=c) * exp(alpha*(cos+mrg)), k=0 slot
  int c = cBase + wave * 16 + ln;
  float s = 0.f;
#pragma unroll
  for (int mt = 0; mt < 8; mt++)
#pragma unroll
    for (int v = 0; v < 8; v++) {
      int m = mt * 16 + v + 8 * kh;                    // C/D VGPR layout
      float cv = acc[mt][v];
      if (Tl[m] != c) s += __expf(ALPHA * (cv + MRG));
    }
  s += __shfl_xor(s, 16, 32);                          // fold lane-halves
  if (kh == 0 && c < NCLS) Nsim[c] = (pNum0[c] != 0) ? s : 0.f;
}

// --- 6. neg contributions of inserted slots (k>=1), vs ORIGINAL P rows -----
__global__ void k_negfix(const int* __restrict__ scanI, const float* __restrict__ P,
                         const float* __restrict__ XnF32, const int* __restrict__ T,
                         float* __restrict__ Nsim) {
  __shared__ float red[256];
  __shared__ float invn_s;
  int cnt = scanI[0];
  int e = blockIdx.x;
  if (e >= cnt) return;
  int c = scanI[2 + e * 3 + 0], slot = scanI[2 + e * 3 + 1];
  size_t r = (size_t)slot * NCLS + c;
  int t = threadIdx.x;
  if (t < 32) {
    float s = 0.f;
    const float* pr = P + r * EMB;
#pragma unroll
    for (int j = 0; j < 8; j++) { float x = pr[t + j * 32]; s += x * x; }
    for (int off = 16; off; off >>= 1) s += __shfl_xor(s, off, 32);
    if (t == 0) invn_s = rsqrtf(s + 1e-12f);
  }
  __syncthreads();
  float contrib = 0.f;
  if (t < BATCH) {
    const float* pr = P + r * EMB;
    const float* xr = XnF32 + t * EMB;
    float d = 0.f;
    for (int eidx = 0; eidx < EMB; eidx++) d += pr[eidx] * xr[eidx];
    if (T[t] != c) contrib = __expf(ALPHA * (d * invn_s + MRG));
  }
  red[t] = contrib;
  __syncthreads();
  for (int s2 = 128; s2 > 0; s2 >>= 1) {
    if (t < s2) red[t] += red[t + s2];
    __syncthreads();
  }
  if (t == 0) atomicAdd(&Nsim[c], red[0]);
}

// --- 7. final loss reduction -----------------------------------------------
__global__ void k_loss(const float* __restrict__ Nsim, const float* __restrict__ scanF,
                       float* __restrict__ out) {
  __shared__ float red[256];
  int t = threadIdx.x;
  float s = 0.f;
  for (int c = t; c < NCLS; c += 256) s += log1pf(Nsim[c]);
  red[t] = s;
  __syncthreads();
  for (int k = 128; k > 0; k >>= 1) {
    if (t < k) red[t] += red[t + k];
    __syncthreads();
  }
  if (t == 0) out[0] = scanF[1] + red[0] / (float)NCLS;
}

// --- 8. bulk P -> P_f copy (dominant HBM cost); b128 loads + prefetch ------
__global__ void k_copyp(const float* __restrict__ P, float* __restrict__ dst) {
  const long total4 = (long)NCLS * KPRX * EMB / 4;  // 6,400,000 float4
  long i = (long)blockIdx.x * 256 + threadIdx.x;
  const long stride = (long)gridDim.x * 256;
  for (; i < total4; i += stride) {
    __builtin_prefetch(((const float4*)P) + i + stride, 0, 0);  // global_prefetch_b8
    float4 v = ((const float4*)P)[i];
    long o = i * 4;            // dst is d_out+1 -> only 4B aligned: b32 stores
    dst[o] = v.x; dst[o + 1] = v.y; dst[o + 2] = v.z; dst[o + 3] = v.w;
  }
}

// --- 9. overwrite inserted proxy rows with raw X rows ----------------------
__global__ void k_fixp(const int* __restrict__ scanI, const float* __restrict__ X,
                       float* __restrict__ dst) {
  int cnt = scanI[0];
  int e = blockIdx.x;
  if (e >= cnt) return;
  int c = scanI[2 + e * 3 + 0], slot = scanI[2 + e * 3 + 1], i = scanI[2 + e * 3 + 2];
  size_t r = (size_t)slot * NCLS + c;
  dst[r * EMB + threadIdx.x] = X[(size_t)i * EMB + threadIdx.x];
}

extern "C" void kernel_launch(void* const* d_in, const int* in_sizes, int n_in,
                              void* d_out, int out_size, void* d_ws, size_t ws_size,
                              hipStream_t stream) {
  const float* X          = (const float*)d_in[0];
  const int* T            = (const int*)d_in[1];
  const float* P          = (const float*)d_in[2];
  const unsigned char* pN = (const unsigned char*)d_in[3];   // bool array
  const int* pTotal       = (const int*)d_in[4];
  float* out = (float*)d_out;

  char* ws = (char*)d_ws;
  float*    XnF32 = (float*)(ws + WS_XN_F32);
  _Float16* XnF16 = (_Float16*)(ws + WS_XN_F16);
  _Float16* PnF16 = (_Float16*)(ws + WS_PN_F16);
  float*    cosT  = (float*)(ws + WS_COST);
  float*    Nsim  = (float*)(ws + WS_NSIM);
  int*      scanI = (int*)(ws + WS_SCAN);
  float*    scanF = (float*)(ws + WS_SCAN);

  k_normx<<<BATCH, EMB, 0, stream>>>(X, XnF32, XnF16);
  k_pnorm<<<NCLS / 8, 256, 0, stream>>>(P, PnF16);
  k_cost<<<(BATCH * KPRX) / 8, 256, 0, stream>>>(P, XnF32, T, cosT);
  k_scan<<<1, 32, 0, stream>>>(T, pTotal, pN, cosT, scanI, scanF);

  size_t smem = (size_t)2 * 128 * 256 * sizeof(_Float16) + 128 * sizeof(int);
  k_neggemm<<<(NCLS + 127) / 128, 256, smem, stream>>>(PnF16, XnF16, T, pN, Nsim);
  k_negfix<<<BATCH, 256, 0, stream>>>(scanI, P, XnF32, T, Nsim);
  k_loss<<<1, 256, 0, stream>>>(Nsim, scanF, out);

  k_copyp<<<4096, 256, 0, stream>>>(P, out + 1);
  k_fixp<<<BATCH, EMB, 0, stream>>>(scanI, X, out + 1);
}
